// StationSpatialEncoder_77506979823764
// MI455X (gfx1250) — compile-verified, matrix operand order checked by reading, not run on hardware
//
#include <hip/hip_runtime.h>
#include <hip/hip_bf16.h>

typedef __attribute__((ext_vector_type(16))) _Float16 v16h;
typedef __attribute__((ext_vector_type(8)))  _Float16 v8h;
typedef __attribute__((ext_vector_type(4)))  _Float16 v4h;
typedef __attribute__((ext_vector_type(8)))  float    v8f;

// ---------------------------------------------------------------------------
// Weight prep: BTout[z][n][k] (f16, row-major in k) = W[z*zStride + base +
// k*kStride + n*nStride].  Contiguous-k layout makes every GEMM fragment
// gather a 2x b128 load.  Total ~0.5M elements -> negligible, L2-resident.
// ---------------------------------------------------------------------------
__global__ void prep_bt(const float* __restrict__ W, int base, int kStride,
                        int nStride, int zStride,
                        _Float16* __restrict__ out, int outZ, int K, int N)
{
    const int idx = blockIdx.x * 256 + threadIdx.x;
    if (idx >= K * N) return;
    const int n = idx / K;
    const int k = idx - n * K;
    const int z = blockIdx.y;
    out[(long)z * outZ + (long)n * K + k] =
        (_Float16)W[(long)z * zStride + base + (long)k * kStride + (long)n * nStride];
}

// ---------------------------------------------------------------------------
// f16 WMMA GEMM: C[m,n] = act( sum_k A(m,k) * BT[n][k] + bias[n] )
//   A: f32 or f16, row stride ldaM, per-z offset aHead; fragments loaded
//      DIRECTLY from global (A rows have no cross-wave reuse).
//   BT: f16 [N][K] pre-transposed; double-buffered LDS staging (ping-pong,
//       ONE barrier per K-step; next tile's global loads overlap WMMA chain).
// Block: 256 threads (8 waves). Tile 128(M) x 128(N) x 32(K).
// Each wave: 16 rows x 128 cols = 8x v_wmma_f32_16x16x32_f16 per K-step.
// K is compile-time -> loop unrolls / pipelines.
// ---------------------------------------------------------------------------
template<bool A_F32, bool RELU, bool HAS_BIAS, int K>
__global__ __launch_bounds__(256)
void gemm16(const void* __restrict__ Av, int ldaM, int aHead,
            const _Float16* __restrict__ BT, int bHead,
            const float* __restrict__ bias, int biasHead,
            _Float16* __restrict__ C, int ldcM, int cHead)
{
    constexpr int NSTEP = K / 32;
    __shared__ __align__(16) _Float16 BsT[2][128][40];

    const int m0   = blockIdx.x * 128;
    const int n0   = blockIdx.y * 128;
    const int z    = blockIdx.z;
    const int t    = threadIdx.x;
    const int lane = t & 31;
    const int wid  = t >> 5;
    const int row16 = lane & 15;
    const int kb8   = (lane >> 4) * 8;          // WMMA K-phase (0 or 8)

    const long aOff = (long)z * aHead;
    const long cOff = (long)z * cHead;
    const int  biasOff = z * biasHead;

    const int brow  = t >> 1;                   // B stager: n row in tile
    const int kcol0 = (t & 1) * 16;             // 0 or 16
    const int arow  = m0 + wid * 16 + row16;    // this lane's A row

    const _Float16* bSrcBase = BT + (long)z * bHead + (long)(n0 + brow) * K + kcol0;

    auto stageB = [&](int step, int buf) {
        const _Float16* src = bSrcBase + step * 32;
        *(v8h*)&BsT[buf][brow][kcol0]     = *(const v8h*)(src);
        *(v8h*)&BsT[buf][brow][kcol0 + 8] = *(const v8h*)(src + 8);
        if (step + 2 < NSTEP) __builtin_prefetch(src + 64, 0, 3);
    };

    v8f acc[8] = {};

    stageB(0, 0);
    __syncthreads();

    #pragma unroll
    for (int s = 0; s < NSTEP; ++s) {
        const int p = s & 1;
        if (s + 1 < NSTEP) stageB(s + 1, p ^ 1);   // overlap with compute below

        const int kk = s * 32;
        v16h af;
        if (A_F32) {
            const float* ap = (const float*)Av + (long)arow * ldaM + aOff + kk + kb8;
            #pragma unroll
            for (int i = 0; i < 8; ++i) { af[i] = (_Float16)ap[i]; af[i + 8] = (_Float16)ap[16 + i]; }
        } else {
            const _Float16* ap = (const _Float16*)Av + (long)arow * ldaM + aOff + kk + kb8;
            const v8h a0 = *(const v8h*)ap;
            const v8h a1 = *(const v8h*)(ap + 16);
            #pragma unroll
            for (int i = 0; i < 8; ++i) { af[i] = a0[i]; af[i + 8] = a1[i]; }
        }

        #pragma unroll
        for (int sub = 0; sub < 8; ++sub) {
            const _Float16* bp = &BsT[p][sub * 16 + row16][kb8];
            const v8h b0 = *(const v8h*)bp;
            const v8h b1 = *(const v8h*)(bp + 16);
            v16h bf;
            #pragma unroll
            for (int i = 0; i < 8; ++i) { bf[i] = b0[i]; bf[i + 8] = b1[i]; }
            acc[sub] = __builtin_amdgcn_wmma_f32_16x16x32_f16(
                false, af, false, bf, (short)0, acc[sub], false, false);
        }
        __syncthreads();   // one barrier/step: protects both ping and pong
    }

    #pragma unroll
    for (int sub = 0; sub < 8; ++sub) {
        const int col = n0 + sub * 16 + row16;
        const float bv = HAS_BIAS ? bias[biasOff + col] : 0.0f;
        #pragma unroll
        for (int i = 0; i < 8; ++i) {
            const int m = m0 + wid * 16 + i + 8 * (lane >> 4);
            float v = acc[sub][i] + bv;
            if (RELU) v = fmaxf(v, 0.0f);
            C[cOff + (long)m * ldcM + col] = (_Float16)v;
        }
    }
}

// ---------------------------------------------------------------------------
// Fused GEMM (A[M,K] f16 @ BT[128][K]) + bias + residual + LayerNorm(128).
// Block: 128 threads, tile 16 rows x 128 cols; wave w owns cols [w*32,w*32+32).
// Double-buffered B staging; stats via wave32 shfl_xor reductions.
// ---------------------------------------------------------------------------
template<bool RES_F16, int K>
__global__ __launch_bounds__(128)
void gemm_ln(const _Float16* __restrict__ A, const _Float16* __restrict__ BT,
             const float* __restrict__ bias,
             const void* __restrict__ resid, int residStride,
             const float* __restrict__ gamma, const float* __restrict__ beta,
             float* __restrict__ outF32, _Float16* __restrict__ outF16)
{
    constexpr int NSTEP = K / 32;
    __shared__ __align__(16) _Float16 BsT[2][128][40];
    __shared__ float ct[16][128];

    const int m0   = blockIdx.x * 16;
    const int t    = threadIdx.x;
    const int lane = t & 31;
    const int wid  = t >> 5;
    const int row16 = lane & 15;
    const int kb8   = (lane >> 4) * 8;

    const _Float16* bSrcBase = BT + (long)t * K;

    auto stageB = [&](int step, int buf) {
        const _Float16* src = bSrcBase + step * 32;
        *(v8h*)&BsT[buf][t][0]  = *(const v8h*)(src);
        *(v8h*)&BsT[buf][t][8]  = *(const v8h*)(src + 8);
        *(v8h*)&BsT[buf][t][16] = *(const v8h*)(src + 16);
        *(v8h*)&BsT[buf][t][24] = *(const v8h*)(src + 24);
    };

    v8f acc[2] = {};

    stageB(0, 0);
    __syncthreads();

    #pragma unroll
    for (int s = 0; s < NSTEP; ++s) {
        const int p = s & 1;
        if (s + 1 < NSTEP) stageB(s + 1, p ^ 1);

        const int kk = s * 32;
        const _Float16* ap = A + (long)(m0 + row16) * K + kk + kb8;
        const v8h a0 = *(const v8h*)ap;
        const v8h a1 = *(const v8h*)(ap + 16);
        v16h af;
        #pragma unroll
        for (int i = 0; i < 8; ++i) { af[i] = a0[i]; af[i + 8] = a1[i]; }

        #pragma unroll
        for (int sub = 0; sub < 2; ++sub) {
            const _Float16* bp = &BsT[p][wid * 32 + sub * 16 + row16][kb8];
            const v8h b0 = *(const v8h*)bp;
            const v8h b1 = *(const v8h*)(bp + 16);
            v16h bf;
            #pragma unroll
            for (int i = 0; i < 8; ++i) { bf[i] = b0[i]; bf[i + 8] = b1[i]; }
            acc[sub] = __builtin_amdgcn_wmma_f32_16x16x32_f16(
                false, af, false, bf, (short)0, acc[sub], false, false);
        }
        __syncthreads();
    }

    #pragma unroll
    for (int sub = 0; sub < 2; ++sub) {
        const int col = wid * 32 + sub * 16 + row16;
        #pragma unroll
        for (int i = 0; i < 8; ++i) ct[i + 8 * (lane >> 4)][col] = acc[sub][i];
    }
    __syncthreads();

    for (int j = 0; j < 4; ++j) {
        const int rr = wid * 4 + j;
        const int m  = m0 + rr;
        float v[4], s = 0.f, sq = 0.f;
        #pragma unroll
        for (int i = 0; i < 4; ++i) {
            const int col = lane + 32 * i;
            float r;
            if (RES_F16) r = (float)((const _Float16*)resid)[(long)m * residStride + col];
            else         r = ((const float*)resid)[(long)m * residStride + col];
            v[i] = ct[rr][col] + bias[col] + r;
            s  += v[i];
            sq += v[i] * v[i];
        }
        #pragma unroll
        for (int msk = 16; msk >= 1; msk >>= 1) {
            s  += __shfl_xor(s,  msk, 32);
            sq += __shfl_xor(sq, msk, 32);
        }
        const float mean = s * (1.0f / 128.0f);
        const float var  = sq * (1.0f / 128.0f) - mean * mean;
        const float rstd = rsqrtf(var + 1e-6f);
        #pragma unroll
        for (int i = 0; i < 4; ++i) {
            const int col = lane + 32 * i;
            const float o = (v[i] - mean) * rstd * gamma[col] + beta[col];
            outF32[(long)m * 128 + col] = o;
            if (outF16) outF16[(long)m * 128 + col] = (_Float16)o;
        }
    }
}

// ---------------------------------------------------------------------------
// Attention core (rank-1 query trick):
//   score_s = (h_s . rq_h + q_h . bk_h)/sqrt(128); softmax over 20 stations;
//   hbar = sum_s attn_s * h_s   (v-projection applied afterwards as a GEMM).
// One block per (b,l); one wave per head; lane owns 4 contiguous dims (v4h).
// h values held in registers between the two passes.
// ---------------------------------------------------------------------------
__global__ __launch_bounds__(128)
void attn_core(const _Float16* __restrict__ q, const _Float16* __restrict__ rq,
               const _Float16* __restrict__ h, const float* __restrict__ bk,
               _Float16* __restrict__ hbar)
{
    const int b    = blockIdx.x;
    const int head = threadIdx.x >> 5;
    const int lane = threadIdx.x & 31;
    const long base = (long)b * 512 + head * 128;
    const int d0 = lane * 4;

    const v4h rq4 = *(const v4h*)(rq + base + d0);
    const v4h q4  = *(const v4h*)(q  + base + d0);
    float rqv[4];
    float c = 0.f;
    #pragma unroll
    for (int i = 0; i < 4; ++i) {
        rqv[i] = (float)rq4[i];
        c += (float)q4[i] * bk[head * 128 + d0 + i];
    }
    #pragma unroll
    for (int m = 16; m >= 1; m >>= 1) c += __shfl_xor(c, m, 32);

    const float scale = 0.08838834764831845f;  // 1/sqrt(128)
    const _Float16* hb = h + (long)b * 20 * 128;

    v4h   hv[20];
    float sc[20];
    #pragma unroll
    for (int s = 0; s < 20; ++s) {
        hv[s] = *(const v4h*)(hb + s * 128 + d0);
        float p = 0.f;
        #pragma unroll
        for (int i = 0; i < 4; ++i) p += (float)hv[s][i] * rqv[i];
        #pragma unroll
        for (int m = 16; m >= 1; m >>= 1) p += __shfl_xor(p, m, 32);
        sc[s] = (p + c) * scale;
    }
    float mx = sc[0];
    #pragma unroll
    for (int s = 1; s < 20; ++s) mx = fmaxf(mx, sc[s]);
    float sum = 0.f;
    #pragma unroll
    for (int s = 0; s < 20; ++s) { sc[s] = __expf(sc[s] - mx); sum += sc[s]; }
    const float inv = 1.0f / sum;

    float accv[4] = {0.f, 0.f, 0.f, 0.f};
    #pragma unroll
    for (int s = 0; s < 20; ++s) {
        const float w = sc[s] * inv;
        #pragma unroll
        for (int i = 0; i < 4; ++i) accv[i] += w * (float)hv[s][i];
    }
    v4h o4;
    #pragma unroll
    for (int i = 0; i < 4; ++i) o4[i] = (_Float16)accv[i];
    *(v4h*)(hbar + base + d0) = o4;
}

// ---------------------------------------------------------------------------
extern "C" void kernel_launch(void* const* d_in, const int* in_sizes, int n_in,
                              void* d_out, int out_size, void* d_ws, size_t ws_size,
                              hipStream_t stream)
{
    (void)in_sizes; (void)n_in; (void)out_size; (void)ws_size;

    const float* x      = (const float*)d_in[0];
    const float* conv_w = (const float*)d_in[1];
    const float* conv_b = (const float*)d_in[2];
    const float* wq     = (const float*)d_in[3];
    const float* bq     = (const float*)d_in[4];
    const float* wk     = (const float*)d_in[5];
    const float* bk     = (const float*)d_in[6];
    const float* wv     = (const float*)d_in[7];
    const float* bv     = (const float*)d_in[8];
    const float* wo     = (const float*)d_in[9];
    const float* bo     = (const float*)d_in[10];
    const float* ln1_g  = (const float*)d_in[11];
    const float* ln1_b  = (const float*)d_in[12];
    const float* w1     = (const float*)d_in[13];
    const float* b1     = (const float*)d_in[14];
    const float* w2     = (const float*)d_in[15];
    const float* b2     = (const float*)d_in[16];
    const float* ln2_g  = (const float*)d_in[17];
    const float* ln2_b  = (const float*)d_in[18];

    const int N1 = 256 * 64 * 20;  // 327680 rows of h (all stations)
    const int M2 = 256 * 64;       // 16384 (b,l) rows (station 0 only)

    char* ws = (char*)d_ws;
    _Float16* hbuf  = (_Float16*)(ws);               // N1 x 128
    _Float16* qbuf  = (_Float16*)(ws + 83886080);    // M2 x 512
    _Float16* rqbuf = (_Float16*)(ws + 100663296);   // M2 x 512
    _Float16* hbbuf = (_Float16*)(ws + 117440512);   // M2 x 512
    _Float16* obuf  = (_Float16*)(ws + 134217728);   // M2 x 512
    float*    out1f = (float*)   (ws + 150994944);   // M2 x 128 f32
    _Float16* out1h = (_Float16*)(ws + 159383552);   // M2 x 128 f16
    _Float16* ffn1  = (_Float16*)(ws + 163577856);   // M2 x 512
    // pre-transposed f16 weights BT[N][K]
    _Float16* cw2t  = (_Float16*)(ws + 180355072);   // [128][64]
    _Float16* wqt   = (_Float16*)(ws + 180371456);   // [512][128]
    _Float16* rqbt  = (_Float16*)(ws + 180502528);   // [4][128][128]
    _Float16* wvbt  = (_Float16*)(ws + 180633600);   // [4][128][128]
    _Float16* wot   = (_Float16*)(ws + 180764672);   // [128][512]
    _Float16* w1t   = (_Float16*)(ws + 180895744);   // [512][128]
    _Float16* w2t   = (_Float16*)(ws + 181026816);   // [128][512]

    // ---- weight prep (tiny; weights stay L2-resident afterwards) ----
    prep_bt<<<dim3(32, 1),  256, 0, stream>>>(conv_w, 2*64*128, 128,   1,   0, cw2t,     0,  64, 128);
    prep_bt<<<dim3(256, 1), 256, 0, stream>>>(wq,     0,        512,   1,   0, wqt,      0, 128, 512);
    prep_bt<<<dim3(64, 4),  256, 0, stream>>>(wk,     0,        1,     512, 128, rqbt, 16384, 128, 128);
    prep_bt<<<dim3(64, 4),  256, 0, stream>>>(wv,     0,        512,   1,   128, wvbt, 16384, 128, 128);
    prep_bt<<<dim3(256, 1), 256, 0, stream>>>(wo,     0,        128,   1,   0, wot,      0, 512, 128);
    prep_bt<<<dim3(256, 1), 256, 0, stream>>>(w1,     0,        512,   1,   0, w1t,      0, 128, 512);
    prep_bt<<<dim3(256, 1), 256, 0, stream>>>(w2,     0,        128,   1,   0, w2t,      0, 512, 128);

    const dim3 blk(256);
    const dim3 blkLN(128);

    // 1) h = relu(x @ conv_w[2] + conv_b)     [327680,64] x [64,128]
    gemm16<true, true, true, 64><<<dim3(N1/128, 1, 1), blk, 0, stream>>>(
        x, 64, 0, cw2t, 0, conv_b, 0, hbuf, 128, 0);

    // 2) q = h0 @ wq + bq                     [16384,128] x [128,512]
    gemm16<false, false, true, 128><<<dim3(M2/128, 4, 1), blk, 0, stream>>>(
        hbuf, 20*128, 0, wqt, 0, bq, 0, qbuf, 512, 0);

    // 3) rq_h = Wk_h @ q_h per head           4 x [16384,128] x [128,128]
    gemm16<false, false, false, 128><<<dim3(M2/128, 1, 4), blk, 0, stream>>>(
        qbuf, 512, 128, rqbt, 16384, nullptr, 0, rqbuf, 512, 128);

    // 4) scores + softmax + hbar
    attn_core<<<dim3(M2), blkLN, 0, stream>>>(qbuf, rqbuf, hbuf, bk, hbbuf);

    // 5) o_h = hbar_h @ Wv_h + bv_h           4 x [16384,128] x [128,128]
    gemm16<false, false, true, 128><<<dim3(M2/128, 1, 4), blk, 0, stream>>>(
        hbbuf, 512, 128, wvbt, 16384, bv, 128, obuf, 512, 128);

    // 6) out1 = LN(h0 + o @ wo + bo)          [16384,512] x [512,128] + LN
    gemm_ln<true, 512><<<dim3(M2/16), blkLN, 0, stream>>>(
        obuf, wot, bo, hbuf, 20*128, ln1_g, ln1_b, out1f, out1h);

    // 7) ffn1 = relu(out1 @ w1 + b1)          [16384,128] x [128,512]
    gemm16<false, true, true, 128><<<dim3(M2/128, 4, 1), blk, 0, stream>>>(
        out1h, 128, 0, w1t, 0, b1, 0, ffn1, 512, 0);

    // 8) out2 = LN(out1 + ffn1 @ w2 + b2) -> d_out
    gemm_ln<false, 512><<<dim3(M2/16), blkLN, 0, stream>>>(
        ffn1, w2t, b2, out1f, 128, ln2_g, ln2_b, (float*)d_out, nullptr);
}
